// MultiheadAttention_77790447665597
// MI455X (gfx1250) — matrix operand order repaired and checked by reading, not compiled, not run on hardware
//
#include <hip/hip_runtime.h>

typedef __attribute__((ext_vector_type(16))) _Float16 v16h;
typedef __attribute__((ext_vector_type(8)))  float    v8f;

#define T_DIM 2048
#define B_DIM 2
#define E_DIM 512
#define H_DIM 8
#define DH    64

// ---------------------------------------------------------------------------
// Kernel 1: fused QKV projection.  qkv[r, j] = x[r, :] . W[j, :] + bias[j]
// r = t*B + b (x is [T,B,E] row-major), j in [0, 3E).
// Emits f16: Q [B,H,T,64] (pre-scaled by Dh^-0.5), K [B,H,T,64], V^T [B,H,64,T].
// Each wave computes one 16x16 tile with v_wmma_f32_16x16x32_f16.
// Two independent accumulators break the WMMA->WMMA RAW hazard chain.
// ---------------------------------------------------------------------------
__global__ __launch_bounds__(256) void qkv_proj_kernel(
    const float* __restrict__ x, const float* __restrict__ w,
    const float* __restrict__ bias,
    _Float16* __restrict__ q, _Float16* __restrict__ k, _Float16* __restrict__ vt)
{
    const int lane = threadIdx.x & 31;
    const int wave = threadIdx.x >> 5;
    const int m0   = blockIdx.y * 16;
    const int n0   = (blockIdx.x * 8 + wave) * 16;
    const int nloc = lane & 15;
    const int hi   = lane >> 4;            // which K half this lane holds

    const float* arow = x + (size_t)(m0 + nloc) * E_DIM;  // A: lane = row
    const float* brow = w + (size_t)(n0 + nloc) * E_DIM;  // B: lane = column

    v8f c0 = {}, c1 = {};
    #pragma unroll 4
    for (int kk = 0; kk < E_DIM; kk += 64) {
        #pragma unroll
        for (int half = 0; half < 2; ++half) {
            const int kb = kk + half * 32;
            v16h a, b;
            const float* pa = arow + kb + hi * 16;    // K 0..15 / 16..31, contiguous
            #pragma unroll
            for (int i = 0; i < 8; ++i) { a[i] = (_Float16)pa[i]; a[i + 8] = (_Float16)pa[i + 8]; }
            const float* pb = brow + kb + hi * 16;    // K 0..15 / 16..31, contiguous
            #pragma unroll
            for (int i = 0; i < 16; ++i) b[i] = (_Float16)pb[i];
            if (half == 0)
                c0 = __builtin_amdgcn_wmma_f32_16x16x32_f16(false, a, false, b, (short)0, c0, false, false);
            else
                c1 = __builtin_amdgcn_wmma_f32_16x16x32_f16(false, a, false, b, (short)0, c1, false, false);
        }
    }

    const int   j     = n0 + nloc;
    const float bj    = bias[j];
    const int   mbase = hi * 8;
    #pragma unroll
    for (int i = 0; i < 8; ++i) {
        const int r  = m0 + mbase + i;            // C: VGPR i -> row mbase+i
        const int t  = r / B_DIM;
        const int bb = r % B_DIM;
        float val = c0[i] + c1[i] + bj;
        if (j < E_DIM) {                          // Q, pre-scale by 1/sqrt(Dh)
            val *= 0.125f;
            const int h = j >> 6, d = j & 63;
            q[(((size_t)bb * H_DIM + h) * T_DIM + t) * DH + d] = (_Float16)val;
        } else if (j < 2 * E_DIM) {               // K
            const int j2 = j - E_DIM, h = j2 >> 6, d = j2 & 63;
            k[(((size_t)bb * H_DIM + h) * T_DIM + t) * DH + d] = (_Float16)val;
        } else {                                  // V stored transposed: [B,H,64,T]
            const int j2 = j - 2 * E_DIM, h = j2 >> 6, d = j2 & 63;
            vt[(((size_t)bb * H_DIM + h) * DH + d) * T_DIM + t] = (_Float16)val;
        }
    }
}

// ---------------------------------------------------------------------------
// Kernel 2: attention.  One WG per (b, 16-row Q tile); wave w == head w.
// Two-pass safe softmax; post-softmax mul_mask; head-summed attn_weights
// accumulated with ds_add_f32 into LDS and written once per 256-wide S chunk.
// ---------------------------------------------------------------------------
__global__ __launch_bounds__(256) void attn_kernel(
    const _Float16* __restrict__ q, const _Float16* __restrict__ k,
    const _Float16* __restrict__ vt,
    const float* __restrict__ add_mask, const float* __restrict__ mul_mask,
    _Float16* __restrict__ attn, float* __restrict__ attnw)
{
    __shared__ float waccum[16][256];   // head-summed probs, one S chunk
    __shared__ float pbuf[8][16][32];   // per-wave prob tile (C->A transpose)
    __shared__ float stats[8][16][2];   // per-head row {max, sumexp}

    const int lane  = threadIdx.x & 31;
    const int wave  = threadIdx.x >> 5;       // == head index
    const int h     = wave;
    const int t0    = blockIdx.x * 16;
    const int b     = blockIdx.y;
    const int nloc  = lane & 15;
    const int hi    = lane >> 4;
    const int mbase = hi * 8;

    for (int i = threadIdx.x; i < 16 * 256; i += 256) ((float*)waccum)[i] = 0.0f;
    __syncthreads();

    const size_t headOff = ((size_t)b * H_DIM + h) * (size_t)T_DIM * DH;
    const _Float16* qh  = q  + headOff;
    const _Float16* kh  = k  + headOff;
    const _Float16* vth = vt + ((size_t)b * H_DIM + h) * (size_t)DH * T_DIM;
    const float* amrow = add_mask + (size_t)b * T_DIM * T_DIM;
    const float* mmrow = mul_mask + (size_t)b * T_DIM * T_DIM;

    // Q fragments (16x64 = two 16x32 A fragments), resident for whole kernel.
    v16h aq0, aq1;
    {
        const _Float16* pq = qh + (size_t)(t0 + nloc) * DH;
        #pragma unroll
        for (int d0 = 0; d0 < 2; ++d0) {
            v16h a;
            const _Float16* p = pq + d0 * 32 + hi * 16;
            #pragma unroll
            for (int i = 0; i < 8; ++i) { a[i] = p[i]; a[i + 8] = p[8 + i]; }
            if (d0 == 0) aq0 = a; else aq1 = a;
        }
    }

    // scores tile: (Q . K^T)[16x16] at column block s0, plus add_mask.
    // Independent accumulators for the two Dh halves (no WMMA RAW chain);
    // merged during the add_mask add.
    auto score_tile = [&](int s0) -> v8f {
        v16h b0, b1;
        const _Float16* pk = kh + (size_t)(s0 + nloc) * DH + hi * 16;
        #pragma unroll
        for (int i = 0; i < 16; ++i) { b0[i] = pk[i]; b1[i] = pk[32 + i]; }
        v8f ca = {}, cb = {};
        ca = __builtin_amdgcn_wmma_f32_16x16x32_f16(false, aq0, false, b0, (short)0, ca, false, false);
        cb = __builtin_amdgcn_wmma_f32_16x16x32_f16(false, aq1, false, b1, (short)0, cb, false, false);
        const float* am = amrow + (size_t)(t0 + mbase) * T_DIM + s0 + nloc;
        v8f c;
        #pragma unroll
        for (int i = 0; i < 8; ++i) c[i] = ca[i] + cb[i] + am[(size_t)i * T_DIM];
        return c;
    };

    // ---- Pass A: per-row running max / sumexp over all S ----
    float m_run = -1e30f, s_run = 0.0f;
    for (int st = 0; st < 128; ++st) {
        const int s0 = st * 16;
        __builtin_prefetch(kh + (size_t)(s0 + 16 + nloc) * DH, 0, 0);
        v8f c = score_tile(s0);
        #pragma unroll
        for (int i = 0; i < 8; ++i) pbuf[wave][mbase + i][nloc] = c[i];
        asm volatile("s_wait_dscnt 0x0" ::: "memory");
        if (lane < 16) {                       // lane j owns row j's stats
            float tmax = pbuf[wave][lane][0];
            #pragma unroll
            for (int n = 1; n < 16; ++n) tmax = fmaxf(tmax, pbuf[wave][lane][n]);
            const float nm  = fmaxf(m_run, tmax);
            float acc = s_run * __expf(m_run - nm);
            #pragma unroll
            for (int n = 0; n < 16; ++n) acc += __expf(pbuf[wave][lane][n] - nm);
            m_run = nm; s_run = acc;
        }
    }
    if (lane < 16) { stats[wave][lane][0] = m_run; stats[wave][lane][1] = s_run; }
    __syncthreads();

    float rmax[8], rinv[8];
    #pragma unroll
    for (int i = 0; i < 8; ++i) {
        rmax[i] = stats[wave][mbase + i][0];
        rinv[i] = 1.0f / stats[wave][mbase + i][1];
    }

    // ---- Pass B: probs, head-sum into waccum, and P.V ----
    v8f cpv[4] = {};                        // 16x64 PV accumulator for this head
    for (int ch = 0; ch < 8; ++ch) {        // 256-wide S chunks
        const int sb = ch * 256;
        for (int pr = 0; pr < 8; ++pr) {    // 32-wide pairs (one PV K step)
            #pragma unroll
            for (int sub = 0; sub < 2; ++sub) {
                const int s0 = sb + pr * 32 + sub * 16;
                __builtin_prefetch(mmrow + (size_t)(t0 + mbase) * T_DIM + s0 + 32 + nloc, 0, 0);
                v8f c = score_tile(s0);
                const float* mm = mmrow + (size_t)(t0 + mbase) * T_DIM + s0 + nloc;
                #pragma unroll
                for (int i = 0; i < 8; ++i) {
                    const float p = __expf(c[i] - rmax[i]) * rinv[i] * mm[(size_t)i * T_DIM];
                    atomicAdd(&waccum[mbase + i][pr * 32 + sub * 16 + nloc], p);  // ds_add_f32
                    pbuf[wave][mbase + i][sub * 16 + nloc] = p;
                }
            }
            asm volatile("s_wait_dscnt 0x0" ::: "memory");
            v16h ap;                         // C-layout -> A-layout via LDS
            {
                const float* pp = &pbuf[wave][nloc][hi * 16];
                #pragma unroll
                for (int i = 0; i < 8; ++i) {
                    ap[i]     = (_Float16)pp[i];
                    ap[i + 8] = (_Float16)pp[8 + i];
                }
            }
            const int sk = sb + pr * 32;
            #pragma unroll
            for (int dt = 0; dt < 4; ++dt) { // V^T rows are contiguous in s
                v16h bv;
                const _Float16* pv = vth + (size_t)(dt * 16 + nloc) * T_DIM + sk + hi * 16;
                #pragma unroll
                for (int i = 0; i < 16; ++i) bv[i] = pv[i];
                cpv[dt] = __builtin_amdgcn_wmma_f32_16x16x32_f16(false, ap, false, bv, (short)0, cpv[dt], false, false);
            }
        }
        __syncthreads();                     // all heads accumulated this chunk
        #pragma unroll 4
        for (int m = 0; m < 16; ++m) {
            const float v = waccum[m][threadIdx.x] * (1.0f / H_DIM);
            attnw[((size_t)b * T_DIM + (t0 + m)) * T_DIM + sb + threadIdx.x] = v;
            waccum[m][threadIdx.x] = 0.0f;
        }
        __syncthreads();
    }

    // PV result -> attn intermediate [T,B,E] (f16); heads own disjoint columns.
    #pragma unroll
    for (int dt = 0; dt < 4; ++dt) {
        #pragma unroll
        for (int i = 0; i < 8; ++i) {
            const int t = t0 + mbase + i;
            const int e = h * DH + dt * 16 + nloc;
            attn[((size_t)t * B_DIM + b) * E_DIM + e] = (_Float16)cpv[dt][i];
        }
    }
}

// ---------------------------------------------------------------------------
// Kernel 3: output projection.  out[r, j] = attn[r, :] . Wo[j, :] + bo[j]
// ---------------------------------------------------------------------------
__global__ __launch_bounds__(256) void out_proj_kernel(
    const _Float16* __restrict__ attn, const float* __restrict__ w,
    const float* __restrict__ bias, float* __restrict__ out)
{
    const int lane = threadIdx.x & 31;
    const int wave = threadIdx.x >> 5;
    const int m0   = blockIdx.y * 16;
    const int n0   = (blockIdx.x * 8 + wave) * 16;
    const int nloc = lane & 15;
    const int hi   = lane >> 4;

    const _Float16* arow = attn + (size_t)(m0 + nloc) * E_DIM;
    const float*    brow = w    + (size_t)(n0 + nloc) * E_DIM;

    v8f c0 = {}, c1 = {};
    #pragma unroll 4
    for (int kk = 0; kk < E_DIM; kk += 64) {
        #pragma unroll
        for (int half = 0; half < 2; ++half) {
            const int kb = kk + half * 32;
            v16h a, b;
            const _Float16* pa = arow + kb + hi * 16;
            #pragma unroll
            for (int i = 0; i < 8; ++i) { a[i] = pa[i]; a[i + 8] = pa[i + 8]; }
            const float* pb = brow + kb + hi * 16;
            #pragma unroll
            for (int i = 0; i < 16; ++i) b[i] = (_Float16)pb[i];
            if (half == 0)
                c0 = __builtin_amdgcn_wmma_f32_16x16x32_f16(false, a, false, b, (short)0, c0, false, false);
            else
                c1 = __builtin_amdgcn_wmma_f32_16x16x32_f16(false, a, false, b, (short)0, c1, false, false);
        }
    }
    const float bj = bias[n0 + nloc];
    #pragma unroll
    for (int i = 0; i < 8; ++i) {
        const int r = m0 + hi * 8 + i;
        out[(size_t)r * E_DIM + n0 + nloc] = c0[i] + c1[i] + bj;
    }
}

// ---------------------------------------------------------------------------
extern "C" void kernel_launch(void* const* d_in, const int* in_sizes, int n_in,
                              void* d_out, int out_size, void* d_ws, size_t ws_size,
                              hipStream_t stream) {
    const float* x    = (const float*)d_in[0];
    const float* am   = (const float*)d_in[1];
    const float* mm   = (const float*)d_in[2];
    const float* wqkv = (const float*)d_in[3];
    const float* bqkv = (const float*)d_in[4];
    const float* wo   = (const float*)d_in[5];
    const float* bo   = (const float*)d_in[6];
    (void)in_sizes; (void)n_in; (void)out_size; (void)ws_size;

    // Workspace layout (16 MB total): Q | K | V^T | attn, all f16.
    char* ws = (char*)d_ws;
    _Float16* q  = (_Float16*)(ws);
    _Float16* k  = (_Float16*)(ws + (size_t)4 * 1024 * 1024);
    _Float16* vt = (_Float16*)(ws + (size_t)8 * 1024 * 1024);
    _Float16* at = (_Float16*)(ws + (size_t)12 * 1024 * 1024);

    float* out   = (float*)d_out;
    float* attnw = out + (size_t)T_DIM * B_DIM * E_DIM;

    // M=4096, N=1536 tiles: 12 x 256 WGs, 8 waves each (one 16x16 tile/wave).
    qkv_proj_kernel<<<dim3(12, 256), 256, 0, stream>>>(x, wqkv, bqkv, q, k, vt);
    // (q-tile, batch) grid; wave == head inside.
    attn_kernel<<<dim3(T_DIM / 16, B_DIM), 256, 0, stream>>>(q, k, vt, am, mm, at, attnw);
    // M=4096, N=512.
    out_proj_kernel<<<dim3(4, 256), 256, 0, stream>>>(at, wo, bo, out);
}
